// SimpleAttention_38019050504981
// MI455X (gfx1250) — matrix operand order repaired and checked by reading, not compiled, not run on hardware
//
#include <hip/hip_runtime.h>
#include <hip/hip_bf16.h>
#include <stdint.h>
#include <type_traits>

// ---------------------------------------------------------------------------
// CDNA5 (gfx1250) fused single-head attention, bf16 WMMA + f32 accumulate,
// async global->LDS tile copies where operands are already bf16.
// ---------------------------------------------------------------------------

typedef __attribute__((ext_vector_type(16))) __bf16 v16bf;
typedef __attribute__((ext_vector_type(8)))  float  v8f;
typedef __attribute__((ext_vector_type(4)))  int    v4i_;

#define BM   128
#define BN   128
#define BKT  32
#define LDSP 40   // 80B row stride: 16B-aligned chunks, 20-dword stride ->
                  // banks 20*m mod 64 distinct for m=0..15 (conflict-free)

#if defined(__has_builtin)
#  if __has_builtin(__builtin_amdgcn_global_load_async_to_lds_b128)
#    define ASYNC_LDS_COPY 1
#  endif
#endif
#ifndef ASYNC_LDS_COPY
#  define ASYNC_LDS_COPY 0
#endif

union FragB { uint32_t u[8]; v16bf v; };

__device__ __forceinline__ void async_b128(const void* g, void* l) {
#if ASYNC_LDS_COPY
    // param0: AS(1) v4i*  (clang printed: "vector_size(4*int) int __device__ *")
    // param1: AS(3) v4i*  (LDS destination), then imm offset, imm cpol
    __builtin_amdgcn_global_load_async_to_lds_b128(
        (__attribute__((address_space(1))) v4i_*)g,
        (__attribute__((address_space(3))) v4i_*)l, 0, 0);
#else
    (void)g; (void)l;
#endif
}
__device__ __forceinline__ void wait_async0() {
#if ASYNC_LDS_COPY
#  if __has_builtin(__builtin_amdgcn_s_wait_asynccnt)
    __builtin_amdgcn_s_wait_asynccnt(0);
#  else
    asm volatile("s_wait_asynccnt 0" ::: "memory");
#  endif
#endif
}

// load 4 contiguous elements from global and produce 4 bf16
__device__ __forceinline__ void g2b4(const float* p, __bf16* o) {
    const float4 f = *(const float4*)p;
    o[0] = (__bf16)f.x; o[1] = (__bf16)f.y; o[2] = (__bf16)f.z; o[3] = (__bf16)f.w;
}
__device__ __forceinline__ void g2b4(const __bf16* p, __bf16* o) {
    union { uint2 u; __bf16 b[4]; } t;
    t.u = *(const uint2*)p;
    o[0] = t.b[0]; o[1] = t.b[1]; o[2] = t.b[2]; o[3] = t.b[3];
}

__device__ __forceinline__ void lds_store4(__bf16* dst, const __bf16* v) {
    union { __bf16 b[2]; uint32_t u; } p0, p1;
    p0.b[0] = v[0]; p0.b[1] = v[1];
    p1.b[0] = v[2]; p1.b[1] = v[3];
    uint32_t* d = (uint32_t*)dst;   // 4B-aligned: even element offset
    d[0] = p0.u; d[1] = p1.u;
}

// C[m][n] = scale * sum_k A[m][k]*B[k][n] + bias[n]
//   B_IS_NK: B stored as [N][K] (e.g. K-matrix for Q*K^T) -> no transpose on fill
//   else   : B stored as [K][N] (weights, V)              -> transpose on fill
template<bool B_IS_NK, bool OUT_BF16, typename TA, typename TB>
__global__ __launch_bounds__(256)
void gemm_bf16_wmma(const TA* __restrict__ A, int lda,
                    const TB* __restrict__ B, int ldb,
                    const float* __restrict__ bias,
                    void* __restrict__ C, int ldc,
                    int K, float scale)
{
    constexpr bool ASYNC_A = (ASYNC_LDS_COPY != 0) && std::is_same<TA, __bf16>::value;
    constexpr bool ASYNC_B = (ASYNC_LDS_COPY != 0) && B_IS_NK && std::is_same<TB, __bf16>::value;

    __shared__ __bf16 sA[BM * LDSP];   // [m][k]
    __shared__ __bf16 sB[BN * LDSP];   // [n][k]  (K-major per output column)

    const int tid  = threadIdx.x;
    const int lane = tid & 31;
    const int wid  = tid >> 5;
    const int wm   = (wid >> 2) * 64;   // wave row base in block tile (0 / 64)
    const int wn   = (wid & 3)  * 32;   // wave col base (0/32/64/96)
    const int h    = lane >> 4;         // lane half (K-split selector)
    const int lr   = lane & 15;

    const long m0 = (long)blockIdx.y * BM;
    const long n0 = (long)blockIdx.x * BN;

    const v8f vzero = {0.f,0.f,0.f,0.f,0.f,0.f,0.f,0.f};
    v8f acc[4][2];
#pragma unroll
    for (int i = 0; i < 4; ++i)
#pragma unroll
        for (int j = 0; j < 2; ++j) acc[i][j] = vzero;

    for (int k0 = 0; k0 < K; k0 += BKT) {
        // ---- fill A tile: BM x BKT ----
        if constexpr (ASYNC_A) {
#pragma unroll
            for (int j = 0; j < 2; ++j) {
                const int c = tid + (j << 8);     // 0..511 chunks of 8 bf16 (16B)
                const int r = c >> 2;             // 0..127
                const int q = (c & 3) << 3;       // 0,8,16,24
                async_b128(&A[(m0 + r) * (long)lda + (k0 + q)], &sA[r * LDSP + q]);
            }
        } else {
#pragma unroll
            for (int j = 0; j < 4; ++j) {
                const int c  = tid + (j << 8);    // 0..1023 chunks of 4
                const int r  = c >> 3;            // 0..127
                const int kk = (c & 7) << 2;      // 0..28
                __bf16 t[4];
                g2b4(&A[(m0 + r) * (long)lda + (k0 + kk)], t);
                lds_store4(&sA[r * LDSP + kk], t);
            }
        }
        // ---- fill B tile as [n][k] ----
        if constexpr (ASYNC_B) {
#pragma unroll
            for (int j = 0; j < 2; ++j) {
                const int c = tid + (j << 8);
                const int n = c >> 2;
                const int q = (c & 3) << 3;
                async_b128(&B[(n0 + n) * (long)ldb + (k0 + q)], &sB[n * LDSP + q]);
            }
        } else if constexpr (B_IS_NK) {
#pragma unroll
            for (int j = 0; j < 4; ++j) {
                const int c  = tid + (j << 8);
                const int n  = c >> 3;
                const int kk = (c & 7) << 2;
                __bf16 t[4];
                g2b4(&B[(n0 + n) * (long)ldb + (k0 + kk)], t);
                lds_store4(&sB[n * LDSP + kk], t);
            }
        } else {
#pragma unroll
            for (int j = 0; j < 4; ++j) {
                const int c = tid + (j << 8);
                const int r = c >> 5;             // 0..31 (k row)
                const int n = (c & 31) << 2;      // 0..124
                __bf16 t[4];
                g2b4(&B[(long)(k0 + r) * ldb + (n0 + n)], t);
                sB[(n + 0) * LDSP + r] = t[0];
                sB[(n + 1) * LDSP + r] = t[1];
                sB[(n + 2) * LDSP + r] = t[2];
                sB[(n + 3) * LDSP + r] = t[3];
            }
        }
        if constexpr (ASYNC_A || ASYNC_B) wait_async0();
        __syncthreads();

        // prefetch next K tile (emits global_prefetch_b8)
        if (k0 + BKT < K) {
            __builtin_prefetch((const void*)&A[(m0 + (tid >> 1)) * (long)lda + (k0 + BKT)], 0, 0);
            if (B_IS_NK)
                __builtin_prefetch((const void*)&B[(n0 + (tid >> 1)) * (long)ldb + (k0 + BKT)], 0, 0);
            else
                __builtin_prefetch((const void*)&B[(long)(k0 + BKT + (tid & 31)) * ldb + n0], 0, 0);
        }

        // ---- build fragments (contiguous k = e + 16*laneHalf in both A and B) and WMMA ----
        FragB fb[2];
#pragma unroll
        for (int nt = 0; nt < 2; ++nt) {
            const int col = wn + nt * 16 + lr;
#pragma unroll
            for (int p = 0; p < 8; ++p)     // B: element e -> K = e + 16*laneHalf
                fb[nt].u[p] = *(const uint32_t*)&sB[col * LDSP + 2 * p + 16 * h];
        }
#pragma unroll
        for (int mt = 0; mt < 4; ++mt) {
            FragB fa;
            const int row = wm + mt * 16 + lr;
#pragma unroll
            for (int p = 0; p < 8; ++p) {   // A: K = 2p + 16*laneHalf
                const int kk = 2 * p + 16 * h;
                fa.u[p] = *(const uint32_t*)&sA[row * LDSP + kk];
            }
#pragma unroll
            for (int nt = 0; nt < 2; ++nt)
                acc[mt][nt] = __builtin_amdgcn_wmma_f32_16x16x32_bf16(
                    false, fa.v, false, fb[nt].v, (short)0, acc[mt][nt], false, false);
        }
        __syncthreads();
    }

    // ---- epilogue: scale + bias, store f32 or bf16 ----
#pragma unroll
    for (int mt = 0; mt < 4; ++mt)
#pragma unroll
        for (int nt = 0; nt < 2; ++nt)
#pragma unroll
            for (int r = 0; r < 8; ++r) {
                const long row = m0 + wm + mt * 16 + r + 8 * h;   // D: M = reg + 8*laneHalf
                const long col = n0 + wn + nt * 16 + lr;          // D: N = lane%16
                float v = acc[mt][nt][r] * scale;
                if (bias) v += bias[col];
                if (OUT_BF16) ((__bf16*)C)[row * (long)ldc + col] = (__bf16)v;
                else          ((float *)C)[row * (long)ldc + col] = v;
            }
}

// Row softmax over n=2048 f32 values, writing bf16 in place (row ld = 2n bf16).
__global__ __launch_bounds__(256)
void softmax_row_to_bf16(float* __restrict__ S, int n)
{
    const int tid = threadIdx.x;
    float* p = S + (long)blockIdx.x * n;

    float v[8];
    float mx = -3.402823e38f;
#pragma unroll
    for (int i = 0; i < 8; ++i) {
        v[i] = p[tid + (i << 8)];
        mx = fmaxf(mx, v[i]);
    }
#pragma unroll
    for (int off = 16; off > 0; off >>= 1)
        mx = fmaxf(mx, __shfl_xor(mx, off, 32));

    __shared__ float red[8];
    if ((tid & 31) == 0) red[tid >> 5] = mx;
    __syncthreads();
    mx = red[0];
#pragma unroll
    for (int j = 1; j < 8; ++j) mx = fmaxf(mx, red[j]);
    __syncthreads();

    float s = 0.f;
#pragma unroll
    for (int i = 0; i < 8; ++i) { v[i] = __expf(v[i] - mx); s += v[i]; }
#pragma unroll
    for (int off = 16; off > 0; off >>= 1)
        s += __shfl_xor(s, off, 32);
    if ((tid & 31) == 0) red[tid >> 5] = s;
    __syncthreads();                       // all f32 reads of this row done here
    float tot = red[0];
#pragma unroll
    for (int j = 1; j < 8; ++j) tot += red[j];
    const float rs = 1.0f / tot;

    __bf16* o = (__bf16*)p;                // bf16 row occupies first half of f32 row
#pragma unroll
    for (int i = 0; i < 8; ++i)
        o[tid + (i << 8)] = (__bf16)(v[i] * rs);
}

extern "C" void kernel_launch(void* const* d_in, const int* in_sizes, int n_in,
                              void* d_out, int out_size, void* d_ws, size_t ws_size,
                              hipStream_t stream)
{
    (void)in_sizes; (void)n_in; (void)out_size; (void)ws_size;

    const float* x  = (const float*)d_in[0];
    const float* Wq = (const float*)d_in[1];
    const float* bq = (const float*)d_in[2];
    const float* Wk = (const float*)d_in[3];
    const float* bk = (const float*)d_in[4];
    const float* Wv = (const float*)d_in[5];
    const float* bv = (const float*)d_in[6];
    const float* Wh = (const float*)d_in[7];
    const float* bh = (const float*)d_in[8];
    float* y = (float*)d_out;

    const int BATCH = 4, S = 2048, H = 1024;
    const int M = BATCH * S;                        // 8192

    // workspace layout (needs 80 MB): Qb|Kb|Vb|Cb bf16 (16MB each) + f32 scores (16MB)
    char* ws = (char*)d_ws;
    __bf16* Qb = (__bf16*)(ws);
    __bf16* Kb = (__bf16*)(ws + ((size_t)16 << 20));
    __bf16* Vb = (__bf16*)(ws + ((size_t)32 << 20));
    __bf16* Cb = (__bf16*)(ws + ((size_t)48 << 20));
    float*  Sc = (float*) (ws + ((size_t)64 << 20));

    const dim3 blk(256);
    const dim3 gproj(H / BN, M / BM);   // (8, 64)
    const dim3 gsc(S / BN, S / BM);     // (16, 16)
    const dim3 gav(H / BN, S / BM);     // (8, 16)

    // 1) QKV projections (f32 in -> bf16 out, bias fused)
    gemm_bf16_wmma<false, true, float, float><<<gproj, blk, 0, stream>>>(
        x, H, Wq, H, bq, Qb, H, H, 1.0f);
    gemm_bf16_wmma<false, true, float, float><<<gproj, blk, 0, stream>>>(
        x, H, Wk, H, bk, Kb, H, H, 1.0f);
    gemm_bf16_wmma<false, true, float, float><<<gproj, blk, 0, stream>>>(
        x, H, Wv, H, bv, Vb, H, H, 1.0f);

    // 2) per-batch attention
    const float scale = 0.03125f;       // 1/sqrt(1024)
    for (int b = 0; b < BATCH; ++b) {
        const __bf16* Qp = Qb + (size_t)b * S * H;
        const __bf16* Kp = Kb + (size_t)b * S * H;
        const __bf16* Vp = Vb + (size_t)b * S * H;
        __bf16*       Cp = Cb + (size_t)b * S * H;

        // scores = scale * Q K^T  (K is [s,h] row-major == [N][K] operand)
        gemm_bf16_wmma<true, false, __bf16, __bf16><<<gsc, blk, 0, stream>>>(
            Qp, H, Kp, H, nullptr, Sc, S, H, scale);

        // softmax rows -> bf16 in place (attn ld = 2S bf16 elements)
        softmax_row_to_bf16<<<S, blk, 0, stream>>>(Sc, S);

        // context = attn @ V
        gemm_bf16_wmma<false, true, __bf16, __bf16><<<gav, blk, 0, stream>>>(
            (const __bf16*)Sc, 2 * S, Vp, H, nullptr, Cp, H, S, 1.0f);
    }

    // 3) y = context @ Wh + bh  (f32 out)
    gemm_bf16_wmma<false, false, __bf16, float><<<gproj, blk, 0, stream>>>(
        Cb, H, Wh, H, bh, y, H, H, 1.0f);
}
